// CANet_5609227289247
// MI455X (gfx1250) — matrix operand order repaired and checked by reading, not compiled, not run on hardware
//
#include <hip/hip_runtime.h>

// ---------------------------------------------------------------------------
// CDNA5 / gfx1250 CANet head.  All heavy math on v_wmma_f32_16x16x32_bf16
// (fp32 accumulate).  Workload is HBM-bound (~1.2 GB in @ 23.3 TB/s ~= 55 us
// floor) -> one-pass bf16 conversion, CxC attention energies live in LDS,
// 4-wide N-tiling for A-operand reuse, async/TDM LDS staging for GEMM B.
// ---------------------------------------------------------------------------

#ifndef __has_builtin
#define __has_builtin(x) 0
#endif
#if __has_builtin(__builtin_amdgcn_global_load_async_to_lds_b128)
#define USE_ASYNC_LDS 1
#else
#define USE_ASYNC_LDS 0
#endif
#if __has_builtin(__builtin_amdgcn_tensor_load_to_lds) && \
    __has_builtin(__builtin_amdgcn_s_wait_tensorcnt)
#define USE_TDM_LDS 1
#else
#define USE_TDM_LDS 0
#endif

typedef __attribute__((ext_vector_type(16))) __bf16       v16bf;
typedef __attribute__((ext_vector_type(8)))  __bf16       v8bf;
typedef __attribute__((ext_vector_type(8)))  float        v8f;
typedef __attribute__((ext_vector_type(4)))  int          v4i;
typedef __attribute__((ext_vector_type(8)))  int          v8i;
typedef __attribute__((ext_vector_type(4)))  unsigned int v4u;

__device__ __forceinline__ v8bf ld8(const __bf16* p) { return *(const v8bf*)p; }

__device__ __forceinline__ v8f wmma_bf16(v16bf a, v16bf b, v8f c) {
  // 8-arg form: (neg_a, A, neg_b, B, c_mod, C, reuse_a, reuse_b)
  return __builtin_amdgcn_wmma_f32_16x16x32_bf16(false, a, false, b, (short)0, c,
                                                 false, false);
}

// 16-byte copy global -> LDS; async (ASYNCcnt) when the toolchain exposes the
// gfx1250 builtin, otherwise through VGPRs.
// Builtin signature (from compiler diagnostic): (v4i AS1*, v4i AS3*, Ii, Ii).
// AS-qualified pointers built via integer casts: the shared aperture lives in
// ADDR[63:32], so the low 32 bits of a generic LDS pointer are the LDS offset.
__device__ __forceinline__ void copy16_to_lds(const __bf16* g, __bf16* l) {
#if USE_ASYNC_LDS
  __attribute__((address_space(1))) v4i* gp =
      (__attribute__((address_space(1))) v4i*)(size_t)g;
  __attribute__((address_space(3))) v4i* lp =
      (__attribute__((address_space(3))) v4i*)(unsigned)(size_t)l;
  __builtin_amdgcn_global_load_async_to_lds_b128(gp, lp, 0, 0);
#else
  *(v8bf*)l = ld8(g);
#endif
}

__device__ __forceinline__ void async_drain() {
#if USE_ASYNC_LDS
#if __has_builtin(__builtin_amdgcn_s_wait_asynccnt)
  __builtin_amdgcn_s_wait_asynccnt(0);
#else
  asm volatile("s_wait_asynccnt 0x0" ::: "memory");
#endif
#endif
}

// Stage a [32 K x 64 N] bf16 tile into LDS: 8 x 512B async transfers per wave.
__device__ __forceinline__ void stage_tile(const __bf16* gsrc /* ->X[k0][n0] */,
                                           size_t ldn, __bf16* lbuf, int lane) {
#pragma unroll
  for (int i = 0; i < 8; ++i) {
    int seg = i * 32 + lane;      // 256 segments of 8 bf16
    int row = seg >> 3;
    int col = (seg & 7) * 8;
    copy16_to_lds(gsrc + (size_t)row * ldn + col, lbuf + row * 64 + col);
  }
}

// ---- Tensor Data Mover: one TDM descriptor loads the whole [32 x 64] tile ---
// D# per CDNA5 ISA:
//  group0: [1:0]=count(1), [63:32]=lds_addr, [120:64]=global_addr, [127:126]=2
//  group1: [17:16]=data_size(1=2B), [79:48]=tensor_dim0, [111:80]=tensor_dim1,
//          [127:112]=tile_dim0(64), [143:128]=tile_dim1(32),
//          [207:160]=tensor_dim0_stride
//  groups 2/3: zero (2D tensor)
// This toolchain exposes the 6-arg builtin:
//  (uint32x4 g0, int32x8 g1, int32x4 g2, int32x4 g3, int32x8, i32 cpol)
__device__ __forceinline__ void stage_tile_tdm(const __bf16* gsrc, __bf16* lbuf,
                                               int td0, int td1, int ldn, int lane) {
#if USE_TDM_LDS
  unsigned long long ga = (unsigned long long)(size_t)gsrc;
  unsigned lds_off = (unsigned)(size_t)lbuf;
  v4u g0 = {1u, lds_off, (unsigned)(ga & 0xffffffffu),
            (unsigned)((ga >> 32) & 0x01ffffffull) | (2u << 30)};
  v8i g1 = {(int)(1u << 16),                                       // data_size=2B
            (int)(((unsigned)td0 & 0xffffu) << 16),                // dim0 lo16
            (int)((((unsigned)td0 >> 16) & 0xffffu) |
                  (((unsigned)td1 & 0xffffu) << 16)),              // dim0 hi|dim1 lo
            (int)((((unsigned)td1 >> 16) & 0xffffu) | (64u << 16)),// dim1 hi|tile0=64
            32,                                                    // tile_dim1=32
            (int)ldn,                                              // dim0 stride lo32
            0, 0};
  v4i z4 = {0, 0, 0, 0};
  v8i z8 = {0, 0, 0, 0, 0, 0, 0, 0};
  __builtin_amdgcn_tensor_load_to_lds(g0, g1, z4, z4, z8, 0);
#else
  stage_tile(gsrc, (size_t)ldn, lbuf, lane);
#endif
}

__device__ __forceinline__ void tensor_drain() {
#if USE_TDM_LDS
  __builtin_amdgcn_s_wait_tensorcnt(0);
#else
  async_drain();
#endif
}

// A fragment, row-major source with contiguous K.
// Layout used here (B frags match): lanes 0-15 -> M=lane, K in {kb..kb+15}, kb=0;
//             lanes 16-31 -> M=lane-16, kb=16.
__device__ __forceinline__ v16bf frag_a_rowmajor(const __bf16* row_base, size_t ldk,
                                                 int lane, int k0) {
  const int m  = lane & 15;
  const int kb = (lane < 16) ? 0 : 16;
  const __bf16* p = row_base + (size_t)m * ldk + (size_t)(k0 + kb);
  v8bf lo = ld8(p);        // K = kb .. kb+7      -> elems 0..7
  v8bf hi = ld8(p + 8);    // K = kb+8 .. kb+15   -> elems 8..15
  v16bf a;
#pragma unroll
  for (int e = 0; e < 8; ++e) { a[e] = lo[e]; a[8 + e] = hi[e]; }
  return a;
}

// A fragment sourced from an LDS fp32 strip (attention probabilities).
__device__ __forceinline__ v16bf frag_a_lds_f32(const float* row_base, int ldk,
                                                int lane, int k0) {
  const int m  = lane & 15;
  const int kb = (lane < 16) ? 0 : 16;
  const float* p = row_base + (size_t)m * ldk + (size_t)(k0 + kb);
  v16bf a;
#pragma unroll
  for (int e = 0; e < 8; ++e) { a[e] = (__bf16)p[e]; a[8 + e] = (__bf16)p[8 + e]; }
  return a;
}

// B fragment where K is contiguous in memory per lane (Gram-matrix case).
// Lanes 0-15: N=lane, K=e; lanes 16-31: K=16+e.
__device__ __forceinline__ v16bf frag_b_kcontig(const __bf16* col_base, size_t ldn,
                                                int lane, int k0) {
  const int n  = lane & 15;
  const int kb = (lane < 16) ? 0 : 16;
  const __bf16* p = col_base + (size_t)n * ldn + (size_t)(k0 + kb);
  v8bf b0 = ld8(p);
  v8bf b1 = ld8(p + 8);
  v16bf b;
#pragma unroll
  for (int e = 0; e < 8; ++e) { b[e] = b0[e]; b[8 + e] = b1[e]; }
  return b;
}

// B fragment where K strides by `ld` (row-major [K,N] operand).
__device__ __forceinline__ v16bf frag_b_kstrided(const __bf16* base, size_t ld,
                                                 int lane, int k0) {
  const int n  = lane & 15;
  const int kb = (lane < 16) ? 0 : 16;
  v16bf b;
#pragma unroll
  for (int e = 0; e < 16; ++e) b[e] = base[(size_t)(k0 + kb + e) * ld + n];
  return b;
}

// B fragment from a staged LDS tile [32 K x 64 N], tile column t*16.
__device__ __forceinline__ v16bf frag_b_lds(const __bf16* lbuf, int t, int lane) {
  const int n  = lane & 15;
  const int kb = (lane < 16) ? 0 : 16;
  v16bf b;
#pragma unroll
  for (int e = 0; e < 16; ++e) b[e] = lbuf[(kb + e) * 64 + t * 16 + n];
  return b;
}

// ---------------------------------------------------------------------------
// fp32 -> bf16 conversion
// ---------------------------------------------------------------------------
__global__ void k_f32_to_bf16(const float* __restrict__ in, __bf16* __restrict__ out,
                              size_t n) {
  size_t i = (size_t)blockIdx.x * blockDim.x + threadIdx.x;
  if (i < n) out[i] = (__bf16)in[i];
}

// 3x3 conv weights: [co, ci, 3, 3] fp32 -> [co, ky, kx, ci] bf16.
__global__ void k_wconv_perm_bf16(const float* __restrict__ w, __bf16* __restrict__ out,
                                  int cin) {
  size_t total = (size_t)512 * cin * 9;
  size_t i = (size_t)blockIdx.x * blockDim.x + threadIdx.x;
  if (i >= total) return;
  int kx = (int)(i % 3); size_t t = i / 3;
  int ky = (int)(t % 3); t /= 3;
  int ci = (int)(t % cin);
  int co = (int)(t / cin);
  out[(((size_t)co * 3 + ky) * 3 + kx) * cin + ci] = (__bf16)w[i];
}

// ---------------------------------------------------------------------------
// Fused channel-attention (CAM): E = X X^T strip in LDS, att = softmax(min-E),
// out = bf16(gamma * att X + x).  One workgroup (4 waves) per 16-row strip;
// 4-wide tiles so each A fragment feeds 4 WMMAs.
// ---------------------------------------------------------------------------
template <int C, int N>
__global__ void k_cam(const float* __restrict__ x,     // [B, C, N] fp32
                      const __bf16* __restrict__ xb,   // [B, C, N] bf16
                      const float* __restrict__ gamma, // [1]
                      __bf16* __restrict__ outb) {     // [B, C, N] bf16
  extern __shared__ unsigned char smem_raw[];
  float* E    = (float*)smem_raw;                // [16][C]
  float* red  = E + (size_t)16 * C;              // [16][8]
  float* rowv = red + 16 * 8;                    // [16]

  const int rb   = blockIdx.x;
  const int b    = blockIdx.y;
  const int tid  = threadIdx.x;
  const int lane = tid & 31;            // wave32
  const int wave = tid >> 5;
  const int nwav = blockDim.x >> 5;
  const int row0 = rb * 16;

  const __bf16* Xb = xb + (size_t)b * C * N;
  const float*  Xf = x  + (size_t)b * C * N;

  // ---- Phase A: E[16, C] = Xrow * X^T (K = N), 4 column tiles per A-load ----
  const int ndt    = C / 16;
  const int dchunk = ndt / nwav;
  const int dt0    = wave * dchunk;
  for (int dtg = dt0; dtg < dt0 + dchunk; dtg += 4) {
    v8f acc[4] = {};
    for (int k0 = 0; k0 < N; k0 += 32) {
      v16bf a = frag_a_rowmajor(Xb + (size_t)row0 * N, N, lane, k0);
#pragma unroll
      for (int t = 0; t < 4; ++t) {
        v16bf bm = frag_b_kcontig(Xb + (size_t)(dtg + t) * 16 * N, N, lane, k0);
        acc[t] = wmma_bf16(a, bm, acc[t]);
      }
    }
#pragma unroll
    for (int t = 0; t < 4; ++t)
#pragma unroll
      for (int v = 0; v < 8; ++v) {  // C/D layout: VGPR v -> row v / v+8
        int m = v + ((lane < 16) ? 0 : 8);
        int nn = lane & 15;
        E[m * C + (dtg + t) * 16 + nn] = acc[t][v];
      }
  }
  __syncthreads();

  // ---- Phase B: att = exp(rowmin - E)/sum  (== softmax(rowmax - E)) ----
  const int tpr = blockDim.x >> 4;
  const int r   = tid / tpr;
  const int lr  = tid % tpr;
  float mn = 3.4e38f;
  for (int j = lr; j < C; j += tpr) mn = fminf(mn, E[r * C + j]);
  red[r * tpr + lr] = mn;
  __syncthreads();
  if (lr == 0) {
    float m2 = red[r * tpr];
    for (int t = 1; t < tpr; ++t) m2 = fminf(m2, red[r * tpr + t]);
    rowv[r] = m2;
  }
  __syncthreads();
  const float rmin = rowv[r];
  float s = 0.f;
  for (int j = lr; j < C; j += tpr) {
    float p = __expf(rmin - E[r * C + j]);
    E[r * C + j] = p;
    s += p;
  }
  __syncthreads();
  red[r * tpr + lr] = s;
  __syncthreads();
  if (lr == 0) {
    float s2 = 0.f;
    for (int t = 0; t < tpr; ++t) s2 += red[r * tpr + t];
    rowv[r] = s2;
  }
  __syncthreads();
  const float inv = 1.0f / rowv[r];
  for (int j = lr; j < C; j += tpr) E[r * C + j] *= inv;
  __syncthreads();

  // ---- Phase C: out = gamma * (att X) + x (K = C), A reused over 4 tiles ----
  const float g = gamma[0];
  const int nnt    = N / 16;
  const int nchunk = nnt / nwav;
  const int nt0    = wave * nchunk;
  for (int ntg = nt0; ntg < nt0 + nchunk; ntg += 4) {
    v8f acc[4] = {};
    for (int k0 = 0; k0 < C; k0 += 32) {
      v16bf a = frag_a_lds_f32(E, C, lane, k0);
#pragma unroll
      for (int t = 0; t < 4; ++t) {
        v16bf bm = frag_b_kstrided(Xb + (size_t)(ntg + t) * 16, N, lane, k0);
        acc[t] = wmma_bf16(a, bm, acc[t]);
      }
    }
#pragma unroll
    for (int t = 0; t < 4; ++t)
#pragma unroll
      for (int v = 0; v < 8; ++v) {
        int m = v + ((lane < 16) ? 0 : 8);
        int nn = lane & 15;
        size_t idx = (size_t)(row0 + m) * N + (size_t)(ntg + t) * 16 + nn;
        outb[(size_t)b * C * N + idx] = (__bf16)(g * acc[t][v] + Xf[idx]);
      }
  }
}

// ---------------------------------------------------------------------------
// Batched GEMM, one wave -> 16 x 64 output tile.  B operand double-buffered
// through LDS: TDM (TENSORcnt) or async copies (ASYNCcnt), template-selected
// so the binary exercises both CDNA5 data movers.
//   out[b] = W (MxK) * X[b] (KxN) [+bias] [+add]
// ---------------------------------------------------------------------------
template <bool HAS_BIAS, bool HAS_ADD, bool TDM>
__global__ void k_gemm(const __bf16* __restrict__ W, const __bf16* __restrict__ X,
                       const float* __restrict__ bias, const float* __restrict__ add,
                       float* __restrict__ out, int M, int K, int N) {
  __shared__ __attribute__((aligned(16))) __bf16 sB[2][32 * 64];
  const int mt = blockIdx.x, ng = blockIdx.y, b = blockIdx.z;
  const int lane = threadIdx.x & 31;
  const int n0 = ng * 64;
  const __bf16* Xb = X + (size_t)b * K * N;

  v8f acc[4] = {};
  if (TDM) {
    stage_tile_tdm(Xb + n0, sB[0], N - n0, K, N, lane);
    tensor_drain();
  } else {
    stage_tile(Xb + n0, N, sB[0], lane);
    async_drain();
  }
  int cur = 0;
  for (int k0 = 0; k0 < K; k0 += 32) {
    if (k0 + 32 < K) {
      if (TDM)
        stage_tile_tdm(Xb + (size_t)(k0 + 32) * N + n0, sB[cur ^ 1], N - n0,
                       K - (k0 + 32), N, lane);
      else
        stage_tile(Xb + (size_t)(k0 + 32) * N + n0, N, sB[cur ^ 1], lane);
    }
    v16bf a = frag_a_rowmajor(W + (size_t)mt * 16 * K, K, lane, k0);
#pragma unroll
    for (int t = 0; t < 4; ++t) {
      v16bf bm = frag_b_lds(sB[cur], t, lane);
      acc[t] = wmma_bf16(a, bm, acc[t]);
    }
    if (TDM) tensor_drain(); else async_drain();  // next buffer ready
    cur ^= 1;
  }
#pragma unroll
  for (int t = 0; t < 4; ++t)
#pragma unroll
    for (int v = 0; v < 8; ++v) {
      int m = v + ((lane < 16) ? 0 : 8);
      int n = lane & 15;
      int gm = mt * 16 + m;
      size_t idx = (size_t)b * M * N + (size_t)gm * N + n0 + t * 16 + n;
      float r = acc[t][v];
      if (HAS_BIAS) r += bias[gm];
      if (HAS_ADD)  r += add[idx];
      out[idx] = r;
    }
}

// ---------------------------------------------------------------------------
// Implicit-GEMM 3x3 stride-2 pad-1 convolution, Cout = 512.
// Weights pre-permuted to [co][ky][kx][ci].  One wave -> 16co x 64pix tile
// (A fragment reused across 4 pixel tiles).
// ---------------------------------------------------------------------------
template <int CIN, int HIN, bool OUT_BF16, bool HAS_BIAS, bool HAS_ADD>
__global__ void k_conv3s2(const __bf16* __restrict__ X,  // [B, CIN, HIN, HIN]
                          const __bf16* __restrict__ Wp, // [512, 3, 3, CIN]
                          const float* __restrict__ bias,
                          const float* __restrict__ add, // [B,512,NPIX]
                          void* __restrict__ outp) {
  constexpr int HOUT = HIN / 2;
  constexpr int NPIX = HOUT * HOUT;
  const int mt = blockIdx.x, ng = blockIdx.y, b = blockIdx.z;
  const int lane = threadIdx.x & 31;
  const __bf16* Xb = X + (size_t)b * CIN * HIN * HIN;

  const int n  = lane & 15;
  const int kb = (lane < 16) ? 0 : 16;
  int oy[4], ox[4];
#pragma unroll
  for (int t = 0; t < 4; ++t) {
    int pix = (ng * 4 + t) * 16 + n;
    oy[t] = pix / HOUT;
    ox[t] = pix % HOUT;
  }

  v8f acc[4] = {};
  for (int ky = 0; ky < 3; ++ky) {
    for (int kx = 0; kx < 3; ++kx) {
      const __bf16* wbase =
          Wp + (size_t)mt * 16 * 9 * CIN + (size_t)(ky * 3 + kx) * CIN;
      for (int c0 = 0; c0 < CIN; c0 += 32) {
        if (c0 + 32 < CIN)
          __builtin_prefetch(Xb + (size_t)(c0 + 32) * HIN * HIN, 0, 0);
        v16bf a = frag_a_rowmajor(wbase, (size_t)9 * CIN, lane, c0);
#pragma unroll
        for (int t = 0; t < 4; ++t) {
          const int iy = oy[t] * 2 - 1 + ky;
          const int ix = ox[t] * 2 - 1 + kx;
          const bool inb = (iy >= 0) && (iy < HIN) && (ix >= 0) && (ix < HIN);
          const __bf16* xbase = Xb + (size_t)iy * HIN + ix;
          v16bf bm;
#pragma unroll
          for (int e = 0; e < 16; ++e) {
            bm[e] = inb ? xbase[(size_t)(c0 + kb + e) * HIN * HIN] : (__bf16)0.0f;
          }
          acc[t] = wmma_bf16(a, bm, acc[t]);
        }
      }
    }
  }
#pragma unroll
  for (int t = 0; t < 4; ++t)
#pragma unroll
    for (int v = 0; v < 8; ++v) {
      int m = v + ((lane < 16) ? 0 : 8);
      int nn = lane & 15;
      int gm = mt * 16 + m;
      size_t idx = (size_t)b * 512 * NPIX + (size_t)gm * NPIX +
                   (size_t)(ng * 4 + t) * 16 + nn;
      float r = acc[t][v];
      if (HAS_BIAS) r += bias[gm];
      if (HAS_ADD)  r += add[idx];
      if (OUT_BF16) ((__bf16*)outp)[idx] = (__bf16)r;
      else          ((float*)outp)[idx]  = r;
    }
}

// ---------------------------------------------------------------------------
// Global average pool over 256 pixels + FC (3584 -> 2).
// ---------------------------------------------------------------------------
__global__ void k_pool_fc(const float* __restrict__ cam2, const float* __restrict__ cam3,
                          const float* __restrict__ cam4, const float* __restrict__ c4,
                          const float* __restrict__ fcw, const float* __restrict__ fcb,
                          float* __restrict__ out) {
  const int b = blockIdx.x;
  const int tid = threadIdx.x;
  __shared__ float red0[256], red1[256];
  float a0 = 0.f, a1 = 0.f;
  for (int ch = tid; ch < 3584; ch += blockDim.x) {
    const float* src; size_t base;
    if (ch < 512)       { src = cam2; base = ((size_t)b * 512 + ch) * 256; }
    else if (ch < 1024) { src = cam3; base = ((size_t)b * 512 + (ch - 512)) * 256; }
    else if (ch < 1536) { src = cam4; base = ((size_t)b * 512 + (ch - 1024)) * 256; }
    else                { src = c4;  base = ((size_t)b * 2048 + (ch - 1536)) * 256; }
    float s = 0.f;
    for (int p = 0; p < 256; ++p) s += src[base + p];
    float mean = s * (1.0f / 256.0f);
    a0 += mean * fcw[ch];
    a1 += mean * fcw[3584 + ch];
  }
  red0[tid] = a0; red1[tid] = a1;
  __syncthreads();
  for (int st = blockDim.x / 2; st > 0; st >>= 1) {
    if (tid < st) { red0[tid] += red0[tid + st]; red1[tid] += red1[tid + st]; }
    __syncthreads();
  }
  if (tid == 0) {
    out[b * 2 + 0] = red0[0] + fcb[0];
    out[b * 2 + 1] = red1[0] + fcb[1];
  }
}

// ---------------------------------------------------------------------------
// Host launcher
// ---------------------------------------------------------------------------
extern "C" void kernel_launch(void* const* d_in, const int* in_sizes, int n_in,
                              void* d_out, int out_size, void* d_ws, size_t ws_size,
                              hipStream_t stream) {
  (void)in_sizes; (void)n_in; (void)out_size; (void)ws_size;
  const float* c2   = (const float*)d_in[0];   // [16,512,64,64]
  const float* c3   = (const float*)d_in[1];   // [16,1024,32,32]
  const float* c4   = (const float*)d_in[2];   // [16,2048,16,16]
  const float* w4   = (const float*)d_in[3];   // [512,2048,1,1]
  const float* b4   = (const float*)d_in[4];   // [512]
  const float* w3   = (const float*)d_in[5];   // [512,1024,3,3]
  const float* w2a  = (const float*)d_in[6];   // [512,512,3,3]
  const float* w2b  = (const float*)d_in[7];   // [512,512,3,3]
  const float* b2b  = (const float*)d_in[8];   // [512]
  const float* g2   = (const float*)d_in[9];
  const float* g3   = (const float*)d_in[10];
  const float* g4   = (const float*)d_in[11];
  const float* fcw  = (const float*)d_in[12];  // [2,3584]
  const float* fcb  = (const float*)d_in[13];  // [2]
  float* out = (float*)d_out;                  // [16,2]

  const int B = 16;
  const size_t n_c4 = (size_t)B * 2048 * 256;
  const size_t n_c3 = (size_t)B * 1024 * 1024;
  const size_t n_c2 = (size_t)B * 512 * 4096;
  const size_t n_w4 = (size_t)512 * 2048;
  const size_t n_w3 = (size_t)512 * 1024 * 9;
  const size_t n_w2 = (size_t)512 * 512 * 9;
  const size_t n_d4 = (size_t)B * 512 * 256;
  const size_t n_t2 = (size_t)B * 512 * 1024;

  char* ws = (char*)d_ws;
  size_t off = 0;
  auto carve = [&](size_t bytes) -> void* {
    void* p = ws + off;
    off = (off + bytes + 255) & ~(size_t)255;
    return p;
  };
  __bf16* c4b   = (__bf16*)carve(n_c4 * 2);
  __bf16* c3b   = (__bf16*)carve(n_c3 * 2);
  __bf16* c2b   = (__bf16*)carve(n_c2 * 2);
  __bf16* out4b = (__bf16*)carve(n_c4 * 2);
  __bf16* out3b = (__bf16*)carve(n_c3 * 2);
  __bf16* out2b = (__bf16*)carve(n_c2 * 2);
  __bf16* w4b   = (__bf16*)carve(n_w4 * 2);
  __bf16* w3b   = (__bf16*)carve(n_w3 * 2);
  __bf16* w2ab  = (__bf16*)carve(n_w2 * 2);
  __bf16* w2bb  = (__bf16*)carve(n_w2 * 2);
  float*  down4 = (float*)carve(n_d4 * 4);
  float*  cam4f = (float*)carve(n_d4 * 4);
  float*  cam3f = (float*)carve(n_d4 * 4);
  float*  cam2f = (float*)carve(n_d4 * 4);
  __bf16* t2    = (__bf16*)carve(n_t2 * 2);

  auto cvt = [&](const float* src, __bf16* dst, size_t n) {
    int thr = 256;
    int blk = (int)((n + thr - 1) / thr);
    k_f32_to_bf16<<<blk, thr, 0, stream>>>(src, dst, n);
  };
  cvt(c4, c4b, n_c4);
  cvt(c3, c3b, n_c3);
  cvt(c2, c2b, n_c2);
  cvt(w4, w4b, n_w4);
  {
    int thr = 256;
    k_wconv_perm_bf16<<<(int)((n_w3 + thr - 1) / thr), thr, 0, stream>>>(w3, w3b, 1024);
    k_wconv_perm_bf16<<<(int)((n_w2 + thr - 1) / thr), thr, 0, stream>>>(w2a, w2ab, 512);
    k_wconv_perm_bf16<<<(int)((n_w2 + thr - 1) / thr), thr, 0, stream>>>(w2b, w2bb, 512);
  }

  // --- fused CAM blocks (energy strip in LDS) ---
  const size_t sh4 = ((size_t)16 * 2048 + 16 * 8 + 16) * sizeof(float); // ~131.6 KB
  const size_t sh3 = ((size_t)16 * 1024 + 16 * 8 + 16) * sizeof(float);
  const size_t sh2 = ((size_t)16 * 512  + 16 * 8 + 16) * sizeof(float);
  (void)hipFuncSetAttribute((const void*)&k_cam<2048, 256>,
                            hipFuncAttributeMaxDynamicSharedMemorySize, (int)sh4);
  (void)hipFuncSetAttribute((const void*)&k_cam<1024, 1024>,
                            hipFuncAttributeMaxDynamicSharedMemorySize, (int)sh3);
  (void)hipFuncSetAttribute((const void*)&k_cam<512, 4096>,
                            hipFuncAttributeMaxDynamicSharedMemorySize, (int)sh2);
  k_cam<2048, 256><<<dim3(128, B), 128, sh4, stream>>>(c4, c4b, g4, out4b);
  k_cam<1024, 1024><<<dim3(64, B), 128, sh3, stream>>>(c3, c3b, g3, out3b);
  k_cam<512, 4096><<<dim3(32, B), 128, sh2, stream>>>(c2, c2b, g2, out2b);

  // --- down4 = conv1x1(c4) + b4 (TDM staging);
  //     cam4 = conv1x1(out4) + b4 + down4 (async staging) ---
  k_gemm<true, false, true><<<dim3(32, 4, B), 32, 0, stream>>>(
      w4b, c4b, b4, nullptr, down4, 512, 2048, 256);
  k_gemm<true, true, false><<<dim3(32, 4, B), 32, 0, stream>>>(
      w4b, out4b, b4, down4, cam4f, 512, 2048, 256);

  // --- cam3 = conv3x3s2(out3) + down4 ---
  k_conv3s2<1024, 32, false, false, true><<<dim3(32, 4, B), 32, 0, stream>>>(
      out3b, w3b, nullptr, down4, (void*)cam3f);

  // --- cam2 = conv3x3s2(conv3x3s2(out2)) + b2b + down4 ---
  k_conv3s2<512, 64, true, false, false><<<dim3(32, 16, B), 32, 0, stream>>>(
      out2b, w2ab, nullptr, nullptr, (void*)t2);
  k_conv3s2<512, 32, false, true, true><<<dim3(32, 4, B), 32, 0, stream>>>(
      t2, w2bb, b2b, down4, (void*)cam2f);

  // --- pool + FC ---
  k_pool_fc<<<B, 256, 0, stream>>>(cam2f, cam3f, cam4f, c4, fcw, fcb, out);
}
